// DeformableAttention_50354196579014
// MI455X (gfx1250) — matrix operand order repaired and checked by reading, not compiled, not run on hardware
//
#include <hip/hip_runtime.h>
#include <hip/hip_bf16.h>
#include <math.h>

// ---------------------------------------------------------------------------
// DeformableAttention on MI455X (gfx1250, wave32, WMMA f32_16x16x32_f16)
// B=8, C=128, H=W=64, G=2, nH=4, dh=32, gc=64, Hk=Wk=16, n=256
// GEMMs register-blocked 16x64 per wave (1 A fragment, 4 B tiles, 16 WMMAs).
// ---------------------------------------------------------------------------

typedef __attribute__((ext_vector_type(16))) _Float16 v16h;
typedef __attribute__((ext_vector_type(8)))  float    v8f;

#define B_   8
#define C_   128
#define H_   64
#define W_   64
#define HW_  4096
#define G_   2
#define NH_  4
#define GC_  64
#define DH_  32
#define NPT_ 256   // Hk*Wk sampled points

// ---- WMMA fragment loaders (16x16x32 f16, wave32 layouts per CDNA5 ISA) ----
// A (16xK row-major): lane l (row = l&15, hi = l>>4):
//   VGPR0-3 <- K = k0 + hi*8 + 0..7 ; VGPR4-7 <- K = k0 + 16 + hi*8 + 0..7
__device__ __forceinline__ v16h load_a_rowmajor(const _Float16* p, int ld,
                                                int row0, int k0, int lane) {
  int r = lane & 15, hi = lane >> 4;
  const _Float16* q = p + (long)(row0 + r) * ld + k0 + hi * 8;
  v16h a;
#pragma unroll
  for (int i = 0; i < 8; i++) { a[i] = q[i]; a[i + 8] = q[i + 16]; }
  return a;
}

// B (Kx16, each column's K values contiguous in memory with stride ld between
// columns): lane l (col = l&15, hi = l>>4) holds K = k0 + hi*8 + 0..7, k0 + 16 + hi*8 + 0..7.
__device__ __forceinline__ v16h load_b_colmajor(const _Float16* p, int ld,
                                                int col0, int k0, int lane) {
  int c = lane & 15, hi = lane >> 4;
  const _Float16* q = p + (long)(col0 + c) * ld + k0 + hi * 8;
  v16h b;
#pragma unroll
  for (int i = 0; i < 8; i++) { b[i] = q[i]; b[i + 8] = q[i + 16]; }
  return b;
}

// Bilinear grid-sample, align_corners=True, zeros padding (torch defaults).
__device__ __forceinline__ float gs_bilinear(const float* img, int Hi, int Wi,
                                             float gy, float gx) {
  float fx = (gx + 1.f) * 0.5f * (float)(Wi - 1);
  float fy = (gy + 1.f) * 0.5f * (float)(Hi - 1);
  float x0f = floorf(fx), y0f = floorf(fy);
  float wx = fx - x0f, wy = fy - y0f;
  int x0 = (int)x0f, y0 = (int)y0f;
  float s = 0.f;
#pragma unroll
  for (int j = 0; j < 2; j++) {
#pragma unroll
    for (int i = 0; i < 2; i++) {
      int xx = x0 + i, yy = y0 + j;
      if ((unsigned)xx < (unsigned)Wi && (unsigned)yy < (unsigned)Hi) {
        float w = (i ? wx : 1.f - wx) * (j ? wy : 1.f - wy);
        s += w * img[yy * Wi + xx];
      }
    }
  }
  return s;
}

// ---------------------------------------------------------------------------
// Kernel 1: convert 4 projection weights (128x128 each, row-major) to f16.
__global__ void k_cvt_w(const float* wq, const float* wk, const float* wv,
                        const float* wo, _Float16* wh) {
  int i = blockIdx.x * 256 + threadIdx.x;
  if (i >= 4 * 16384) return;
  const float* s = (i < 16384) ? wq : (i < 32768) ? wk : (i < 49152) ? wv : wo;
  wh[i] = (_Float16)s[i & 16383];
}

// Kernel 2: x NCHW fp32 -> channel-last f16 (pixel rows, 128 contiguous ch).
__global__ void k_cvt_x(const float* x, _Float16* xt) {
  int i = blockIdx.x * 256 + threadIdx.x;
  if (i >= B_ * C_ * HW_) return;
  int hw = i & (HW_ - 1);
  int t = i >> 12;
  int c = t & (C_ - 1);
  int b = t >> 7;
  xt[((long)b * HW_ + hw) * C_ + c] = (_Float16)x[i];
}

// Kernel 3: q = wq * x. WMMA GEMM, M=128(out ch), K=128, N=B*HW pixels.
// Wave computes a 16x64 block: 1 A fragment shared by 4 B tiles per K-step.
// Outputs q channel-last fp32 (for conv_offset) and f16 (for attention).
__global__ __launch_bounds__(256) void k_gemm_q(const _Float16* w,
                                                const _Float16* xt,
                                                float* qf, _Float16* qh) {
  int lane = threadIdx.x & 31, wave = threadIdx.x >> 5;
  int job = blockIdx.x * 8 + wave;             // 8 Mtiles x 512 Nquads
  int mt = job >> 9, nq = job & 511;
  int n0 = nq * 64;
  v8f acc[4] = {};
#pragma unroll
  for (int k0 = 0; k0 < 128; k0 += 32) {
    v16h a = load_a_rowmajor(w, 128, mt * 16, k0, lane);
    if (k0 < 96)   // prefetch next K-step's B columns (global_prefetch_b8)
      __builtin_prefetch(xt + (long)(n0 + (lane & 15)) * 128 + k0 + 32, 0, 1);
#pragma unroll
    for (int t = 0; t < 4; t++) {
      v16h b = load_b_colmajor(xt, 128, n0 + t * 16, k0, lane);
      acc[t] = __builtin_amdgcn_wmma_f32_16x16x32_f16(false, a, false, b,
                                                      (short)0, acc[t],
                                                      false, false);
    }
  }
  int cc = lane & 15, hi = lane >> 4;
#pragma unroll
  for (int t = 0; t < 4; t++) {
#pragma unroll
    for (int r = 0; r < 8; r++) {
      int ch = mt * 16 + r + hi * 8;
      long pix = (long)(n0 + t * 16) + cc;
      qf[pix * 128 + ch] = acc[t][r];
      qh[pix * 128 + ch] = (_Float16)acc[t][r];
    }
  }
}

// Kernel 4: depthwise 5x5 stride 4 pad 2 on q (per group), from channel-last q.
__global__ void k_dwconv(const float* qf, const float* wdw, float* off0) {
  int i = blockIdx.x * 256 + threadIdx.x;      // 16*64*256
  if (i >= 16 * 64 * 256) return;
  int p = i & 255;
  int t = i >> 8;
  int c = t & 63;
  int bg = t >> 6;
  int b = bg >> 1, g = bg & 1;
  int ky = p >> 4, kx = p & 15;
  float s = 0.f;
#pragma unroll
  for (int dy = 0; dy < 5; dy++) {
    int y = ky * 4 - 2 + dy;
    if ((unsigned)y >= 64u) continue;
#pragma unroll
    for (int dx = 0; dx < 5; dx++) {
      int x = kx * 4 - 2 + dx;
      if ((unsigned)x >= 64u) continue;
      s += wdw[c * 25 + dy * 5 + dx] *
           qf[((long)b * HW_ + y * 64 + x) * 128 + g * 64 + c];
    }
  }
  off0[((long)bg * 64 + c) * 256 + p] = s;
}

// Kernel 5: BiasFree-LN over channels -> GELU -> 1x1 to 3 -> tanh/sigmoid ->
// pos (y,x in [-1,1], ref grid added) and modulation dm.
__global__ void k_offset(const float* off0, const float* lnw,
                         const float* woff, float* pos, float* dm) {
  int i = blockIdx.x * 64 + threadIdx.x;       // 16*256
  if (i >= 16 * 256) return;
  int p = i & 255, bg = i >> 8;
  float g[64];
  float mean = 0.f;
#pragma unroll
  for (int c = 0; c < 64; c++) {
    float v = off0[((long)bg * 64 + c) * 256 + p];
    g[c] = v; mean += v;
  }
  mean *= (1.f / 64.f);
  float m2 = 0.f;
#pragma unroll
  for (int c = 0; c < 64; c++) { float d = g[c] - mean; m2 += d * d; }
  float inv = rsqrtf(m2 * (1.f / 64.f) + 1e-5f);
  float o0 = 0.f, o1 = 0.f, o2 = 0.f;
#pragma unroll
  for (int c = 0; c < 64; c++) {
    float v = g[c] * inv * lnw[c];
    v = 0.5f * v * (1.f + erff(v * 0.70710678f));   // exact GELU
    o0 += woff[0 * 64 + c] * v;
    o1 += woff[1 * 64 + c] * v;
    o2 += woff[2 * 64 + c] * v;
  }
  int ky = p >> 4, kx = p & 15;
  float refy = ((ky + 0.5f) / 16.f) * 2.f - 1.f;
  float refx = ((kx + 0.5f) / 16.f) * 2.f - 1.f;
  pos[((long)bg * 256 + p) * 2 + 0] = tanhf(o0) * (1.f / 16.f) + refy;
  pos[((long)bg * 256 + p) * 2 + 1] = tanhf(o1) * (1.f / 16.f) + refx;
  dm[bg * 256 + p] = 1.f / (1.f + expf(-o2));
}

// Kernel 6: grid-sample x at pos, modulate, store channel-last f16 (B*n, 128).
__global__ void k_sample(const float* x, const float* pos, const float* dm,
                         _Float16* xsh) {
  int i = blockIdx.x * 256 + threadIdx.x;      // 16*64*256
  if (i >= 16 * 64 * 256) return;
  int p = i & 255;
  int t = i >> 8;
  int c = t & 63;
  int bg = t >> 6;
  int b = bg >> 1, g = bg & 1;
  int ci = g * 64 + c;
  float py = pos[((long)bg * 256 + p) * 2 + 0];
  float px = pos[((long)bg * 256 + p) * 2 + 1];
  const float* img = x + ((long)b * 128 + ci) * HW_;
  float v = gs_bilinear(img, 64, 64, py, px) * dm[bg * 256 + p];
  xsh[((long)b * 256 + p) * 128 + ci] = (_Float16)v;
}

// Kernel 7: k/v projections from xs, 16x64 per wave. vmode=0 -> k channel-last
// (B*n,128); vmode=1 -> v channel-major (B,128,n) for contiguous PV fragments.
__global__ __launch_bounds__(256) void k_gemm_kv(const _Float16* w,
                                                 const _Float16* xsh,
                                                 _Float16* outh, int vmode) {
  int lane = threadIdx.x & 31, wave = threadIdx.x >> 5;
  int job = blockIdx.x * 8 + wave;             // 8 Mtiles x 32 Nquads
  int mt = job >> 5, nq = job & 31;
  int n0 = nq * 64;
  v8f acc[4] = {};
#pragma unroll
  for (int k0 = 0; k0 < 128; k0 += 32) {
    v16h a = load_a_rowmajor(w, 128, mt * 16, k0, lane);
#pragma unroll
    for (int t = 0; t < 4; t++) {
      v16h b = load_b_colmajor(xsh, 128, n0 + t * 16, k0, lane);
      acc[t] = __builtin_amdgcn_wmma_f32_16x16x32_f16(false, a, false, b,
                                                      (short)0, acc[t],
                                                      false, false);
    }
  }
  int cc = lane & 15, hi = lane >> 4;
#pragma unroll
  for (int t = 0; t < 4; t++) {
#pragma unroll
    for (int r = 0; r < 8; r++) {
      int ch = mt * 16 + r + hi * 8;
      int pix = n0 + t * 16 + cc;               // b*256 + p
      if (!vmode) {
        outh[(long)pix * 128 + ch] = (_Float16)acc[t][r];
      } else {
        int b = pix >> 8, p = pix & 255;
        outh[((long)b * 128 + ch) * 256 + p] = (_Float16)acc[t][r];
      }
    }
  }
}

// Kernel 8: fused attention: logits (WMMA) + sampled RPE bias + softmax + PV
// (WMMA). One block = 32 queries (2 waves x 16), one (b, head).
__global__ __launch_bounds__(64) void k_attn(const _Float16* qh,
                                             const _Float16* kh,
                                             const _Float16* vh,
                                             const float* pos,
                                             const float* rpe,
                                             _Float16* oat) {
  __shared__ float    logits[32][256];
  __shared__ _Float16 probs[32][256];
  __shared__ float    poss[256][2];

  int lane = threadIdx.x & 31, wave = threadIdx.x >> 5;
  int blk = blockIdx.x;                        // B*NH*128
  int qb = blk & 127;
  int t = blk >> 7;
  int h = t & 3;
  int b = t >> 2;
  int bg = b * 2 + (h >> 1);

  for (int j = threadIdx.x; j < 256; j += 64) {
    poss[j][0] = pos[((long)bg * 256 + j) * 2 + 0];
    poss[j][1] = pos[((long)bg * 256 + j) * 2 + 1];
  }
  __syncthreads();

  int q0 = qb * 32 + wave * 16;                // query pixel base
  const _Float16* qbase = qh + ((long)b * HW_ + q0) * 128 + h * 32;
  v16h afrag = load_a_rowmajor(qbase, 128, 0, 0, lane);
  const float* rimg = rpe + (long)h * 127 * 127;
  const float scale = 0.17677669529663687f;    // 32^-0.5
  int hi = lane >> 4, cc = lane & 15;

  // --- logits + bias over 16 key tiles (Q fragment reused for all 16) ---
  for (int kt = 0; kt < 16; kt++) {
    v16h bfrag = load_b_colmajor(kh + (long)b * 256 * 128 + h * 32,
                                 128, kt * 16, 0, lane);
    v8f acc = {};
    acc = __builtin_amdgcn_wmma_f32_16x16x32_f16(false, afrag, false, bfrag,
                                                 (short)0, acc, false, false);
    int n = kt * 16 + cc;
    float pyk = poss[n][0], pxk = poss[n][1];
#pragma unroll
    for (int r = 0; r < 8; r++) {
      int m = r + hi * 8;
      int pix = q0 + m;
      int qy = pix >> 6, qx = pix & 63;
      float qgy = ((qy + 0.5f) / 64.f) * 2.f - 1.f;
      float qgx = ((qx + 0.5f) / 64.f) * 2.f - 1.f;
      float dy = (qgy - pyk) * 0.5f;
      float dx = (qgx - pxk) * 0.5f;
      float bias = gs_bilinear(rimg, 127, 127, dy, dx);
      logits[wave * 16 + m][n] = acc[r] * scale + bias;
    }
  }
  __syncthreads();

  // --- softmax over 256 keys, one row per iteration, wave32 reductions ---
  for (int rr = 0; rr < 16; rr++) {
    float* row = logits[wave * 16 + rr];
    float vals[8];
    float mx = -1e30f;
#pragma unroll
    for (int j = 0; j < 8; j++) { vals[j] = row[lane + 32 * j]; mx = fmaxf(mx, vals[j]); }
#pragma unroll
    for (int o = 16; o > 0; o >>= 1) mx = fmaxf(mx, __shfl_xor(mx, o, 32));
    float sum = 0.f;
#pragma unroll
    for (int j = 0; j < 8; j++) { vals[j] = __expf(vals[j] - mx); sum += vals[j]; }
#pragma unroll
    for (int o = 16; o > 0; o >>= 1) sum += __shfl_xor(sum, o, 32);
    float rinv = 1.f / sum;
#pragma unroll
    for (int j = 0; j < 8; j++)
      probs[wave * 16 + rr][lane + 32 * j] = (_Float16)(vals[j] * rinv);
  }
  __syncthreads();

  // --- PV: (16 x 256) x (256 x 32) via 2 column tiles, 8 K-steps each ---
  const _Float16* pr = &probs[wave * 16][0];
#pragma unroll
  for (int ct = 0; ct < 2; ct++) {
    v8f acc = {};
    const _Float16* vbase = vh + ((long)b * 128 + h * 32 + ct * 16) * 256;
#pragma unroll
    for (int k0 = 0; k0 < 256; k0 += 32) {
      v16h a = load_a_rowmajor(pr, 256, 0, k0, lane);
      v16h bv = load_b_colmajor(vbase, 256, 0, k0, lane);
      acc = __builtin_amdgcn_wmma_f32_16x16x32_f16(false, a, false, bv,
                                                   (short)0, acc, false, false);
    }
#pragma unroll
    for (int r = 0; r < 8; r++) {
      int m = r + hi * 8;
      int ch = h * 32 + ct * 16 + cc;
      oat[((long)b * HW_ + q0 + m) * 128 + ch] = (_Float16)acc[r];
    }
  }
}

// Kernel 9: final projection wo * out_attn -> NCHW fp32, 16x64 per wave.
__global__ __launch_bounds__(256) void k_gemm_o(const _Float16* w,
                                                const _Float16* oat,
                                                float* out) {
  int lane = threadIdx.x & 31, wave = threadIdx.x >> 5;
  int job = blockIdx.x * 8 + wave;             // 8 Mtiles x 512 Nquads
  int mt = job >> 9, nq = job & 511;
  int n0 = nq * 64;
  v8f acc[4] = {};
#pragma unroll
  for (int k0 = 0; k0 < 128; k0 += 32) {
    v16h a = load_a_rowmajor(w, 128, mt * 16, k0, lane);
    if (k0 < 96)
      __builtin_prefetch(oat + (long)(n0 + (lane & 15)) * 128 + k0 + 32, 0, 1);
#pragma unroll
    for (int t = 0; t < 4; t++) {
      v16h b = load_b_colmajor(oat, 128, n0 + t * 16, k0, lane);
      acc[t] = __builtin_amdgcn_wmma_f32_16x16x32_f16(false, a, false, b,
                                                      (short)0, acc[t],
                                                      false, false);
    }
  }
  int cc = lane & 15, hi = lane >> 4;
#pragma unroll
  for (int t = 0; t < 4; t++) {
#pragma unroll
    for (int r = 0; r < 8; r++) {
      int ch = mt * 16 + r + hi * 8;
      int pix = n0 + t * 16 + cc;
      int b = pix >> 12, hw = pix & 4095;
      out[((long)b * 128 + ch) * HW_ + hw] = acc[t][r];
    }
  }
}

// ---------------------------------------------------------------------------
extern "C" void kernel_launch(void* const* d_in, const int* in_sizes, int n_in,
                              void* d_out, int out_size, void* d_ws,
                              size_t ws_size, hipStream_t stream) {
  (void)in_sizes; (void)n_in; (void)out_size; (void)ws_size;
  const float* x    = (const float*)d_in[0];
  const float* wdw  = (const float*)d_in[1];
  const float* lnw  = (const float*)d_in[2];
  const float* woff = (const float*)d_in[3];
  const float* wq   = (const float*)d_in[4];
  const float* wk   = (const float*)d_in[5];
  const float* wv   = (const float*)d_in[6];
  const float* wo   = (const float*)d_in[7];
  const float* rpe  = (const float*)d_in[8];
  float* out = (float*)d_out;

  char* ws = (char*)d_ws;
  _Float16* xt  = (_Float16*)ws; ws += (size_t)B_ * HW_ * C_ * 2;   // 8 MB
  _Float16* wh  = (_Float16*)ws; ws += (size_t)4 * 16384 * 2;       // 128 KB
  float*    qf  = (float*)ws;    ws += (size_t)B_ * HW_ * C_ * 4;   // 16 MB
  _Float16* qh  = (_Float16*)ws; ws += (size_t)B_ * HW_ * C_ * 2;   // 8 MB
  float*    off0= (float*)ws;    ws += (size_t)16 * 64 * 256 * 4;   // 1 MB
  float*    pos = (float*)ws;    ws += (size_t)16 * 256 * 2 * 4;    // 32 KB
  float*    dm  = (float*)ws;    ws += (size_t)16 * 256 * 4;        // 16 KB
  _Float16* xsh = (_Float16*)ws; ws += (size_t)B_ * 256 * C_ * 2;   // 512 KB
  _Float16* khh = (_Float16*)ws; ws += (size_t)B_ * 256 * C_ * 2;   // 512 KB
  _Float16* vhh = (_Float16*)ws; ws += (size_t)B_ * 256 * C_ * 2;   // 512 KB
  _Float16* oat = (_Float16*)ws; ws += (size_t)B_ * HW_ * C_ * 2;   // 8 MB

  _Float16* wqh = wh;
  _Float16* wkh = wh + 16384;
  _Float16* wvh = wh + 32768;
  _Float16* woh = wh + 49152;

  k_cvt_w<<<(4 * 16384 + 255) / 256, 256, 0, stream>>>(wq, wk, wv, wo, wh);
  k_cvt_x<<<(B_ * C_ * HW_) / 256, 256, 0, stream>>>(x, xt);
  k_gemm_q<<<512, 256, 0, stream>>>(wqh, xt, qf, qh);
  k_dwconv<<<(16 * 64 * 256) / 256, 256, 0, stream>>>(qf, wdw, off0);
  k_offset<<<(16 * 256) / 64, 64, 0, stream>>>(off0, lnw, woff, pos, dm);
  k_sample<<<(16 * 64 * 256) / 256, 256, 0, stream>>>(x, pos, dm, xsh);
  k_gemm_kv<<<32, 256, 0, stream>>>(wkh, xsh, khh, 0);
  k_gemm_kv<<<32, 256, 0, stream>>>(wvh, xsh, vhh, 1);
  k_attn<<<B_ * NH_ * 128, 64, 0, stream>>>(qh, khh, vhh, pos, rpe, oat);
  k_gemm_o<<<512, 256, 0, stream>>>(woh, oat, out);
}
